// ScalelessViT_21483426414706
// MI455X (gfx1250) — matrix operand order repaired and checked by reading, not compiled, not run on hardware
//
#include <hip/hip_runtime.h>
#include <hip/hip_bf16.h>

// ---------------------------------------------------------------------------
// ScalelessViT forward for gfx1250 (MI455X): f16 WMMA GEMMs + f32 VALU glue.
// ---------------------------------------------------------------------------

typedef __attribute__((ext_vector_type(16))) _Float16 v16h;
typedef __attribute__((ext_vector_type(8)))  _Float16 v8h;
typedef __attribute__((ext_vector_type(8)))  float    v8f;

#define BB      256
#define CW      512
#define D0      64
#define D1      64
#define LATENT  1024
#define HIST    8
#define HEADS   8
#define DIMH    64
#define INNER   512
#define TOK     (BB * HIST)      // 2048 token rows
#define NCLS    1000

// ---------------- crop + resample + ResNet gain ----------------------------
__global__ void __launch_bounds__(256)
crop_kernel(const float* __restrict__ x, const float* __restrict__ tf,
            float* __restrict__ feats)
{
    __shared__ int xi[D0];
    __shared__ int yi[D1];
    const int b   = blockIdx.x;
    const int tid = threadIdx.x;
    const float px = tf[b * 4 + 0], py = tf[b * 4 + 1];
    const float zx = tf[b * 4 + 2], zy = tf[b * 4 + 3];
    if (tid < 64) {
        float x1 = px * 448.0f;
        float xs = fminf(512.0f - x1, zx * 448.0f + 64.0f);
        int v = (int)(floorf(((float)tid * (1.0f / 64.0f)) * xs) + x1);
        xi[tid] = v < 0 ? 0 : (v > 511 ? 511 : v);
    } else if (tid < 128) {
        int j = tid - 64;
        float y1 = py * 448.0f;
        float ys = fminf(512.0f - y1, zy * 448.0f + 64.0f);
        int v = (int)(floorf(((float)j * (1.0f / 64.0f)) * ys) + y1);
        yi[j] = v < 0 ? 0 : (v > 511 ? 511 : v);
    }
    __syncthreads();
    const float* img = x + (size_t)b * CW * CW;
    float* o = feats + (size_t)b * (D0 * D1);
    #pragma unroll
    for (int q = 0; q < 16; ++q) {
        int idx = tid + q * 256;
        int i = idx >> 6, j = idx & 63;
        o[idx] = img[xi[i] * CW + yi[j]] * 128.0f;   // RESNET_GAIN = 2^7
    }
}

// ---------------- build z = [history[1:], image_latent] ---------------------
__global__ void __launch_bounds__(256)
assemble_z(const float* __restrict__ hist, const float* __restrict__ latent,
           float* __restrict__ z)
{
    int idx = blockIdx.x * 256 + threadIdx.x;       // over B*HIST*LATENT
    int k = idx & (LATENT - 1);
    int r = idx >> 10;                              // b*HIST + t
    int t = r & (HIST - 1);
    int b = r >> 3;
    float v = (t < HIST - 1)
            ? hist[((size_t)(t + 1) * BB + b) * LATENT + k]
            : latent[(size_t)b * LATENT + k];
    z[idx] = v;
}

// ---------------- LayerNorm: one wave32 per 1024-wide row -------------------
__global__ void __launch_bounds__(256)
ln_kernel(const float* __restrict__ X, const float* __restrict__ sc,
          const float* __restrict__ bi, float* __restrict__ Y)
{
    const int row  = blockIdx.x * 8 + (threadIdx.x >> 5);
    const int lane = threadIdx.x & 31;
    const float* xr = X + (size_t)row * LATENT;
    float vals[32];
    float sum = 0.f, sq = 0.f;
    #pragma unroll
    for (int q = 0; q < 32; ++q) {
        float v = xr[lane + q * 32];
        vals[q] = v; sum += v; sq += v * v;
    }
    #pragma unroll
    for (int o = 16; o > 0; o >>= 1) {
        sum += __shfl_xor(sum, o, 32);
        sq  += __shfl_xor(sq,  o, 32);
    }
    float mean = sum * (1.0f / LATENT);
    float var  = sq  * (1.0f / LATENT) - mean * mean;
    float inv  = rsqrtf(var + 1e-5f);
    float* yr = Y + (size_t)row * LATENT;
    #pragma unroll
    for (int q = 0; q < 32; ++q) {
        int k = lane + q * 32;
        yr[k] = (vals[q] - mean) * inv * sc[k] + bi[k];
    }
}

// ---------------- tiny attention (seq=8, head=64): f32 VALU ----------------
__global__ void __launch_bounds__(256)
attn_kernel(const float* __restrict__ qkv, float* __restrict__ out)
{
    const int t = blockIdx.x * 256 + threadIdx.x;   // 0 .. B*HEADS*HIST-1
    const int b   = t >> 6;
    const int rem = t & 63;
    const int hh  = rem >> 3;
    const int i   = rem & 7;
    const size_t rs = 3 * INNER;                    // 1536
    const float* qp = qkv + (size_t)(b * HIST + i) * rs + hh * DIMH;
    float q[DIMH];
    #pragma unroll
    for (int e = 0; e < DIMH; ++e) q[e] = qp[e];

    float s[HIST];
    float mx = -1e30f;
    #pragma unroll
    for (int j = 0; j < HIST; ++j) {
        const float* kp = qkv + (size_t)(b * HIST + j) * rs + INNER + hh * DIMH;
        float d = 0.f;
        #pragma unroll
        for (int e = 0; e < DIMH; ++e) d += q[e] * kp[e];
        s[j] = d * 0.125f;                          // dim_head^-0.5
        mx = fmaxf(mx, s[j]);
    }
    float sum = 0.f;
    #pragma unroll
    for (int j = 0; j < HIST; ++j) { s[j] = expf(s[j] - mx); sum += s[j]; }
    const float inv = 1.0f / sum;

    float acc[DIMH];
    #pragma unroll
    for (int e = 0; e < DIMH; ++e) acc[e] = 0.f;
    #pragma unroll
    for (int j = 0; j < HIST; ++j) {
        float p = s[j] * inv;
        const float* vp = qkv + (size_t)(b * HIST + j) * rs + 2 * INNER + hh * DIMH;
        #pragma unroll
        for (int e = 0; e < DIMH; ++e) acc[e] += p * vp[e];
    }
    float* op = out + (size_t)(b * HIST + i) * INNER + hh * DIMH;
    #pragma unroll
    for (int e = 0; e < DIMH; ++e) op[e] = acc[e];
}

// ---------------- WMMA GEMM: C = act(A@B + bias) (+resid) ------------------
// A [M,K] f32 row-major, B [K,N] f32 row-major; converted to f16 in LDS.
// 256 threads = 8 waves; macro tile 64x64; K-step 32; wave -> one 16-row
// strip x two 16-col subtiles (2 x v_wmma_f32_16x16x32_f16 per K-step).
// N and K are compile-time so the 8 strided B loads fold their q*N offsets
// into instruction immediates. M must be a multiple of 64 (all call sites).
// NGUARD=true only for the N=1000 classifier head.
#define TM 64
#define TN 64
#define TK 32
#define LP 40   // padded halfs per LDS row (80 B, keeps 16-B alignment)

template<int ACT, bool NGUARD, bool HB, bool HR, int N, int K>
__global__ void __launch_bounds__(256)
gemm_wmma(const float* __restrict__ A, const float* __restrict__ Bw,
          const float* __restrict__ bias, const float* __restrict__ resid,
          float* __restrict__ Cout)
{
    __shared__ _Float16 As[TM][LP];
    __shared__ _Float16 Bs[TN][LP];   // stored transposed: Bs[n][k]

    const int tid  = threadIdx.x;
    const int lane = tid & 31;
    const int wave = tid >> 5;
    const int mBase = blockIdx.y * TM;
    const int nBase = blockIdx.x * TN;
    const int m0 = (wave & 3) * 16;
    const int n0 = (wave >> 2) * 32;

    v8f acc0 = {}; v8f acc1 = {};

    // A staging: thread -> (row, 8 consecutive k)
    const int arow = tid >> 2;            // 0..63
    const int acol = (tid & 3) * 8;       // 0,8,16,24
    // B staging: thread -> (one n column, 8 consecutive k)  => single b128
    // LDS store; global reads are dword, coalesced across lanes per k.
    const int bn = tid & 63;              // n within tile
    const int bk = (tid >> 6) * 8;        // k base: 0,8,16,24
    const bool bok = !NGUARD || (nBase + bn) < N;

    // Strength-reduced global pointers: one 64-bit add per tile per iter;
    // per-load offsets q*N are compile-time immediates.
    const float* aptr = A + (size_t)(mBase + arow) * K + acol;
    const float* bptr = Bw + (size_t)bk * N + nBase + bn;

    // fragment geometry (ISA 7.12.2, wave32)
    const int fr    = m0 + (lane & 15);   // A row in tile
    const int abase = (lane >> 4) * 16;   // A K base
    const int kb    = (lane >> 4) * 16;   // B K base
    const int fc    = lane & 15;          // B/C column

    for (int k0 = 0; k0 < K; k0 += TK) {
        // stage A tile (f32 -> f16): 2x global b128, 1x ds b128
        {
            _Float16 ta[8];
            #pragma unroll
            for (int q = 0; q < 8; ++q) ta[q] = (_Float16)aptr[q];
            *(v8h*)&As[arow][acol] = *(const v8h*)ta;
            aptr += TK;
        }
        // stage B tile transposed (f32 -> f16): 8x dword clause, 1x ds b128
        {
            _Float16 tb[8];
            if (bok) {
                #pragma unroll
                for (int q = 0; q < 8; ++q) tb[q] = (_Float16)bptr[q * N];
            } else {
                #pragma unroll
                for (int q = 0; q < 8; ++q) tb[q] = (_Float16)0.0f;
            }
            *(v8h*)&Bs[bn][bk] = *(const v8h*)tb;
            bptr += (size_t)TK * N;
            if (k0 + TK < K)  // emits global_prefetch_b8 on next tile
                __builtin_prefetch(bptr, 0, 0);
        }
        __syncthreads();

        // A fragment: lane holds row fr, K = abase .. abase+15 contiguous
        v8h alo = *(const v8h*)&As[fr][abase];
        v8h ahi = *(const v8h*)&As[fr][abase + 8];
        v16h a;
        #pragma unroll
        for (int j = 0; j < 8; ++j) { a[j] = alo[j]; a[j + 8] = ahi[j]; }

        // B fragments: lane holds col, K = kb .. kb+15 contiguous
        {
            v8h bl = *(const v8h*)&Bs[n0 + fc][kb];
            v8h bh = *(const v8h*)&Bs[n0 + fc][kb + 8];
            v16h bfr;
            #pragma unroll
            for (int j = 0; j < 8; ++j) { bfr[j] = bl[j]; bfr[j + 8] = bh[j]; }
            acc0 = __builtin_amdgcn_wmma_f32_16x16x32_f16(
                       false, a, false, bfr, (short)0, acc0, false, false);
        }
        {
            v8h bl = *(const v8h*)&Bs[n0 + 16 + fc][kb];
            v8h bh = *(const v8h*)&Bs[n0 + 16 + fc][kb + 8];
            v16h bfr;
            #pragma unroll
            for (int j = 0; j < 8; ++j) { bfr[j] = bl[j]; bfr[j + 8] = bh[j]; }
            acc1 = __builtin_amdgcn_wmma_f32_16x16x32_f16(
                       false, a, false, bfr, (short)0, acc1, false, false);
        }
        __syncthreads();
    }

    // epilogue: C[m][n], lane -> col fc, VGPR v -> row v + 8*(lane>>4)
    const int mrow = mBase + m0 + 8 * (lane >> 4);
    #pragma unroll
    for (int half = 0; half < 2; ++half) {
        const v8f& acc = half ? acc1 : acc0;
        const int n = nBase + n0 + half * 16 + fc;
        if (!NGUARD || n < N) {
            float bv = HB ? bias[n] : 0.0f;
            #pragma unroll
            for (int v = 0; v < 8; ++v) {
                int m = mrow + v;
                float val = acc[v] + bv;
                if (ACT == 1)
                    val = 0.5f * val * (1.0f + erff(val * 0.70710678118654752f));
                if (HR) val += resid[(size_t)m * N + n];
                Cout[(size_t)m * N + n] = val;
            }
        }
    }
}

// ---------------- transform head: sigmoid(tl @ w_tr + b_tr) ----------------
__global__ void __launch_bounds__(256)
transform_kernel(const float* __restrict__ z, const float* __restrict__ w,
                 const float* __restrict__ bias, float* __restrict__ out)
{
    __shared__ float red[4][256];
    const int b = blockIdx.x, tid = threadIdx.x;
    const float* row = z + (size_t)b * (HIST * LATENT);
    float p0 = 0.f, p1 = 0.f, p2 = 0.f, p3 = 0.f;
    for (int k = tid; k < HIST * LATENT; k += 256) {
        float v = row[k];
        p0 += v * w[k * 4 + 0]; p1 += v * w[k * 4 + 1];
        p2 += v * w[k * 4 + 2]; p3 += v * w[k * 4 + 3];
    }
    red[0][tid] = p0; red[1][tid] = p1; red[2][tid] = p2; red[3][tid] = p3;
    __syncthreads();
    for (int s = 128; s > 0; s >>= 1) {
        if (tid < s) {
            #pragma unroll
            for (int j = 0; j < 4; ++j) red[j][tid] += red[j][tid + s];
        }
        __syncthreads();
    }
    if (tid < 4)
        out[b * 4 + tid] = 1.0f / (1.0f + expf(-(red[tid][0] + bias[tid])));
}

// ---------------------------------------------------------------------------
extern "C" void kernel_launch(void* const* d_in, const int* in_sizes, int n_in,
                              void* d_out, int out_size, void* d_ws, size_t ws_size,
                              hipStream_t stream)
{
    const float* x     = (const float*)d_in[0];
    const float* trf   = (const float*)d_in[1];
    const float* hist  = (const float*)d_in[2];
    const float* w_img = (const float*)d_in[3];
    const float* b_img = (const float*)d_in[4];
    const float* ln1_s = (const float*)d_in[5];
    const float* ln1_b = (const float*)d_in[6];
    const float* w_qkv = (const float*)d_in[7];
    const float* w_o   = (const float*)d_in[8];
    const float* b_o   = (const float*)d_in[9];
    const float* ln2_s = (const float*)d_in[10];
    const float* ln2_b = (const float*)d_in[11];
    const float* w_ff1 = (const float*)d_in[12];
    const float* b_ff1 = (const float*)d_in[13];
    const float* w_ff2 = (const float*)d_in[14];
    const float* b_ff2 = (const float*)d_in[15];
    const float* w_cls = (const float*)d_in[16];
    const float* b_cls = (const float*)d_in[17];
    const float* w_tr  = (const float*)d_in[18];
    const float* b_tr  = (const float*)d_in[19];
    (void)in_sizes; (void)n_in; (void)out_size; (void)ws_size;

    float* ws     = (float*)d_ws;
    float* feats  = ws;  ws += (size_t)BB * D0 * D1;        // 256x4096
    float* latent = ws;  ws += (size_t)BB * LATENT;         // 256x1024
    float* z      = ws;  ws += (size_t)TOK * LATENT;        // 2048x1024
    float* h      = ws;  ws += (size_t)TOK * LATENT;        // 2048x1024
    float* qkv    = ws;  ws += (size_t)TOK * 3 * INNER;     // 2048x1536
    float* attno  = ws;  ws += (size_t)TOK * INNER;         // 2048x512
    float* g      = ws;  ws += (size_t)TOK * LATENT;        // 2048x1024

    // 1) crop/resample + gain
    crop_kernel<<<BB, 256, 0, stream>>>(x, trf, feats);

    // 2) image projection: [256,4096] @ [4096,1024] + b_img
    gemm_wmma<0, false, true, false, LATENT, D0 * D1>
        <<<dim3(LATENT / TN, BB / TM), 256, 0, stream>>>(
            feats, w_img, b_img, nullptr, latent);

    // 3) z = concat(history[1:], latent)  -> [2048,1024]
    assemble_z<<<(TOK * LATENT) / 256, 256, 0, stream>>>(hist, latent, z);

    // 4) transformer layers
    for (int i = 0; i < 4; ++i) {
        ln_kernel<<<TOK / 8, 256, 0, stream>>>(z, ln1_s + i * LATENT,
                                               ln1_b + i * LATENT, h);
        gemm_wmma<0, false, false, false, 3 * INNER, LATENT>
            <<<dim3(3 * INNER / TN, TOK / TM), 256, 0, stream>>>(
                h, w_qkv + (size_t)i * LATENT * 3 * INNER, nullptr, nullptr, qkv);
        attn_kernel<<<(BB * HEADS * HIST) / 256, 256, 0, stream>>>(qkv, attno);
        gemm_wmma<0, false, true, true, LATENT, INNER>
            <<<dim3(LATENT / TN, TOK / TM), 256, 0, stream>>>(
                attno, w_o + (size_t)i * INNER * LATENT, b_o + i * LATENT, z, z);
        ln_kernel<<<TOK / 8, 256, 0, stream>>>(z, ln2_s + i * LATENT,
                                               ln2_b + i * LATENT, h);
        gemm_wmma<1, false, true, false, LATENT, LATENT>
            <<<dim3(LATENT / TN, TOK / TM), 256, 0, stream>>>(
                h, w_ff1 + (size_t)i * LATENT * LATENT, b_ff1 + i * LATENT,
                nullptr, g);
        gemm_wmma<0, false, true, true, LATENT, LATENT>
            <<<dim3(LATENT / TN, TOK / TM), 256, 0, stream>>>(
                g, w_ff2 + (size_t)i * LATENT * LATENT, b_ff2 + i * LATENT, z, z);
    }

    // 5) heads: z (contiguous) == tl [256, 8192]
    float* cls_out = (float*)d_out;
    gemm_wmma<0, true, true, false, NCLS, HIST * LATENT>
        <<<dim3((NCLS + TN - 1) / TN, BB / TM), 256, 0, stream>>>(
            z, w_cls, b_cls, nullptr, cls_out);
    transform_kernel<<<BB, 256, 0, stream>>>(z, w_tr, b_tr, cls_out + BB * NCLS);
}
